// InteractionBlock_48902497632897
// MI455X (gfx1250) — compile-verified, matrix operand order checked by reading, not run on hardware
//
#include <hip/hip_runtime.h>
#include <hip/hip_bf16.h>

#define N_ATOMS_C 50000
#define N_EDGES_C 800000
#define NB 128   // atom basis
#define NF 128   // filters
#define NG 64    // gaussians
#define CUTOFF_F 5.0f

typedef __attribute__((ext_vector_type(16))) __bf16 v16bf;
typedef __attribute__((ext_vector_type(8)))  __bf16 v8bf;
typedef __attribute__((ext_vector_type(8)))  float  v8f;

__device__ __forceinline__ __bf16 f2bf(float f) { return (__bf16)f; }

__device__ __forceinline__ v8f v8f_zero() {
  v8f z;
#pragma unroll
  for (int i = 0; i < 8; ++i) z[i] = 0.0f;
  return z;
}

__device__ __forceinline__ v8f wmma_bf16(v16bf a, v16bf b, v8f c) {
  return __builtin_amdgcn_wmma_f32_16x16x32_bf16(false, a, false, b, (short)0, c,
                                                 false, false);
}

__device__ __forceinline__ v16bf join8(v8bf lo, v8bf hi) {
  return __builtin_shufflevector(lo, hi, 0, 1, 2, 3, 4, 5, 6, 7,
                                 8, 9, 10, 11, 12, 13, 14, 15);
}

// ---- A fragment (16x32 bf16 M x K) from row-major bf16, rows clamped ----
// Lanes 0-15: M=lane, K = {k0+0..7, k0+16..23}
// Lanes 16-31: M=lane-16, K = {k0+8..15, k0+24..31}
__device__ __forceinline__ v16bf a_frag_bf16(const __bf16* __restrict__ base,
                                             int row0, int maxrow, int ld, int k0) {
  int lane = threadIdx.x & 31;
  int m = row0 + (lane & 15);
  if (m > maxrow) m = maxrow;
  int kh = (lane >> 4) ? 8 : 0;
  const __bf16* p = base + (unsigned)m * ld + k0 + kh;  // 16B aligned
  v8bf lo = *(const v8bf*)(p);
  v8bf hi = *(const v8bf*)(p + 16);
  return join8(lo, hi);
}

// ---- A fragment from f32 row-major: vector loads + packed convert ----
__device__ __forceinline__ v16bf a_frag_f32(const float* __restrict__ base,
                                            int row0, int maxrow, int ld, int k0) {
  int lane = threadIdx.x & 31;
  int m = row0 + (lane & 15);
  if (m > maxrow) m = maxrow;
  int kh = (lane >> 4) ? 8 : 0;
  const float* p = base + (unsigned)m * ld + k0 + kh;   // 32B aligned
  v8f lo = *(const v8f*)(p);
  v8f hi = *(const v8f*)(p + 16);
  v8bf lob = __builtin_convertvector(lo, v8bf);
  v8bf hib = __builtin_convertvector(hi, v8bf);
  return join8(lob, hib);
}

// ---- A fragment from bf16 LDS tile (row-major M x ld, 16B-aligned rows) ----
__device__ __forceinline__ v16bf a_frag_lds(const __bf16* base, int ld, int k0) {
  int lane = threadIdx.x & 31;
  int m = lane & 15;
  int kh = (lane >> 4) ? 8 : 0;
  const __bf16* p = base + m * ld + k0 + kh;
  v8bf lo = *(const v8bf*)(p);
  v8bf hi = *(const v8bf*)(p + 16);
  return join8(lo, hi);
}

// ---- B fragment from fragment-packed bf16: [frag][lane][16] contiguous ----
__device__ __forceinline__ v16bf b_frag_pk(const __bf16* __restrict__ packed,
                                           int frag) {
  int lane = threadIdx.x & 31;
  return *(const v16bf*)(packed + (unsigned)(frag * 32 + lane) * 16);  // 32B aligned
}

// ------------------------------------------------------------------
// Kernel: zero fill
// ------------------------------------------------------------------
__global__ void k_zero(float* __restrict__ p, long n) {
  long i = (long)blockIdx.x * blockDim.x + threadIdx.x;
  if (i < n) p[i] = 0.0f;
}

// ------------------------------------------------------------------
// Kernel: elementwise f32 -> bf16 (r matrix, one-time)
// ------------------------------------------------------------------
__global__ void k_cvt(const float* __restrict__ src, __bf16* __restrict__ dst,
                      long n) {
  long i = (long)blockIdx.x * blockDim.x + threadIdx.x;
  if (i < n) dst[i] = f2bf(src[i]);
}

// ------------------------------------------------------------------
// Kernel: pack a [K x 128] f32 weight into B-fragment-native bf16 layout.
// ------------------------------------------------------------------
__global__ __launch_bounds__(32) void k_pack_w(const float* __restrict__ src,
                                               __bf16* __restrict__ dst) {
  int lane = threadIdx.x;
  int kt = blockIdx.x >> 3;
  int nt = blockIdx.x & 7;
  int n = nt * 16 + (lane & 15);
  int kh = (lane >> 4) ? 8 : 0;
  int k0 = kt * 32;
  __bf16* o = dst + (unsigned)(blockIdx.x * 32 + lane) * 16;
#pragma unroll
  for (int i = 0; i < 8; ++i) o[i] = f2bf(src[(unsigned)(k0 + kh + i) * NF + n]);
#pragma unroll
  for (int i = 0; i < 8; ++i)
    o[8 + i] = f2bf(src[(unsigned)(k0 + 16 + kh + i) * NF + n]);
}

// ------------------------------------------------------------------
// Kernel 1: rf = r @ Wa   (bf16 in, f32 out)
// ------------------------------------------------------------------
__global__ __launch_bounds__(256) void k_atomfilter(const __bf16* __restrict__ rbf,
                                                    const __bf16* __restrict__ WaP,
                                                    float* __restrict__ rf) {
  int wave = threadIdx.x >> 5;
  int lane = threadIdx.x & 31;
  int row0 = blockIdx.x * 128 + wave * 16;
  if (row0 >= N_ATOMS_C) return;  // wave-uniform
  v8f acc[8];
#pragma unroll
  for (int t = 0; t < 8; ++t) acc[t] = v8f_zero();
#pragma unroll
  for (int kt = 0; kt < 4; ++kt) {
    v16bf af = a_frag_bf16(rbf, row0, N_ATOMS_C - 1, NB, kt * 32);
#pragma unroll
    for (int nt = 0; nt < 8; ++nt)
      acc[nt] = wmma_bf16(af, b_frag_pk(WaP, kt * 8 + nt), acc[nt]);
  }
  int n = lane & 15;
  int mofs = (lane >> 4) * 8;
#pragma unroll
  for (int nt = 0; nt < 8; ++nt) {
#pragma unroll
    for (int i = 0; i < 8; ++i) {
      int row = row0 + i + mofs;
      if (row < N_ATOMS_C) rf[(unsigned)row * NF + nt * 16 + n] = acc[nt][i];
    }
  }
}

// ------------------------------------------------------------------
// Kernel 2: Gaussian smearing -> g@Wf2 + bf2 -> * rf[src] -> atomic add yacc[dst]
// ------------------------------------------------------------------
__global__ __launch_bounds__(256) void k_edges(const float* __restrict__ e,
                                               const int* __restrict__ a2,
                                               const __bf16* __restrict__ Wf2P,
                                               const float* __restrict__ bf2,
                                               const float* __restrict__ rf,
                                               float* __restrict__ yacc) {
  __shared__ __bf16 gtile[8][16][NG];  // 16 KB
  int wave = threadIdx.x >> 5;
  int lane = threadIdx.x & 31;
  unsigned e0 = blockIdx.x * 128 + wave * 16;

  const float width = CUTOFF_F / (float)(NG - 1);
  const float coeff = -0.5f / (width * width);

#pragma unroll
  for (int idx = lane; idx < 16 * NG; idx += 32) {
    int m = idx >> 6;
    int j = idx & 63;
    float ev = e[e0 + m];
    float d = ev - (float)j * width;
    gtile[wave][m][j] = f2bf(__expf(coeff * d * d));
  }
  __syncthreads();

  v8f acc[8];
#pragma unroll
  for (int t = 0; t < 8; ++t) acc[t] = v8f_zero();
#pragma unroll
  for (int kt = 0; kt < 2; ++kt) {
    v16bf af = a_frag_lds(&gtile[wave][0][0], NG, kt * 32);
#pragma unroll
    for (int nt = 0; nt < 8; ++nt)
      acc[nt] = wmma_bf16(af, b_frag_pk(Wf2P, kt * 8 + nt), acc[nt]);
  }

  int n = lane & 15;
  int mofs = (lane >> 4) * 8;
  unsigned srco[8], dsto[8];
#pragma unroll
  for (int i = 0; i < 8; ++i) {
    unsigned edge = e0 + i + mofs;
    dsto[i] = (unsigned)a2[2 * edge] * NF;      // row byte-offset base (elems)
    srco[i] = (unsigned)a2[2 * edge + 1] * NF;
  }
#pragma unroll
  for (int nt = 0; nt < 8; ++nt) {
    unsigned col = nt * 16 + n;
    float bias = bf2[col];
#pragma unroll
    for (int i = 0; i < 8; ++i) {
      float w = acc[nt][i] + bias;
      float val = rf[srco[i] + col] * w;
      atomicAdd(&yacc[dsto[i] + col], val);
    }
  }
}

// ------------------------------------------------------------------
// Kernel 3: out = ssp(yacc@W1 + b1) @ W2 + b2
// ------------------------------------------------------------------
__global__ __launch_bounds__(256) void k_dense(const float* __restrict__ yacc,
                                               const __bf16* __restrict__ W1P,
                                               const float* __restrict__ b1,
                                               const __bf16* __restrict__ W2P,
                                               const float* __restrict__ b2,
                                               float* __restrict__ out) {
  __shared__ __bf16 htile[8][16][NB];  // 32 KB
  int wave = threadIdx.x >> 5;
  int lane = threadIdx.x & 31;
  int row0 = blockIdx.x * 128 + wave * 16;
  bool active = row0 < N_ATOMS_C;  // wave-uniform

  int n = lane & 15;
  int mofs = (lane >> 4) * 8;

  if (active) {
    v8f acc[8];
#pragma unroll
    for (int t = 0; t < 8; ++t) acc[t] = v8f_zero();
#pragma unroll
    for (int kt = 0; kt < 4; ++kt) {
      v16bf af = a_frag_f32(yacc, row0, N_ATOMS_C - 1, NB, kt * 32);
#pragma unroll
      for (int nt = 0; nt < 8; ++nt)
        acc[nt] = wmma_bf16(af, b_frag_pk(W1P, kt * 8 + nt), acc[nt]);
    }
#pragma unroll
    for (int nt = 0; nt < 8; ++nt) {
      float bias = b1[nt * 16 + n];
#pragma unroll
      for (int i = 0; i < 8; ++i) {
        float x = acc[nt][i] + bias;
        float s = fmaxf(x, 0.0f) + log1pf(__expf(-fabsf(x))) - 0.69314718f;
        htile[wave][i + mofs][nt * 16 + n] = f2bf(s);
      }
    }
  }
  __syncthreads();
  if (active) {
    v8f acc2[8];
#pragma unroll
    for (int t = 0; t < 8; ++t) acc2[t] = v8f_zero();
#pragma unroll
    for (int kt = 0; kt < 4; ++kt) {
      v16bf af = a_frag_lds(&htile[wave][0][0], NB, kt * 32);
#pragma unroll
      for (int nt = 0; nt < 8; ++nt)
        acc2[nt] = wmma_bf16(af, b_frag_pk(W2P, kt * 8 + nt), acc2[nt]);
    }
#pragma unroll
    for (int nt = 0; nt < 8; ++nt) {
      float bias = b2[nt * 16 + n];
#pragma unroll
      for (int i = 0; i < 8; ++i) {
        int row = row0 + i + mofs;
        if (row < N_ATOMS_C)
          out[(unsigned)row * NB + nt * 16 + n] = acc2[nt][i] + bias;
      }
    }
  }
}

// ------------------------------------------------------------------
extern "C" void kernel_launch(void* const* d_in, const int* in_sizes, int n_in,
                              void* d_out, int out_size, void* d_ws, size_t ws_size,
                              hipStream_t stream) {
  const float* r   = (const float*)d_in[0];
  const float* e   = (const float*)d_in[1];
  const float* Wf2 = (const float*)d_in[2];
  const float* bf2 = (const float*)d_in[3];
  const float* Wa  = (const float*)d_in[4];
  const float* W1  = (const float*)d_in[5];
  const float* b1  = (const float*)d_in[6];
  const float* W2  = (const float*)d_in[7];
  const float* b2  = (const float*)d_in[8];
  const int*   a2  = (const int*)d_in[9];
  float* out = (float*)d_out;

  // workspace layout
  float*  rf   = (float*)d_ws;                              // [N,128] f32
  float*  yacc = rf + (size_t)N_ATOMS_C * NF;               // [N,128] f32
  __bf16* rbf  = (__bf16*)(yacc + (size_t)N_ATOMS_C * NF);  // [N,128] bf16
  __bf16* WaP  = rbf + (size_t)N_ATOMS_C * NB;              // 128x128 packed
  __bf16* Wf2P = WaP + NB * NF;                             // 64x128 packed
  __bf16* W1P  = Wf2P + NG * NF;                            // 128x128 packed
  __bf16* W2P  = W1P + NF * NB;                             // 128x128 packed

  long ycount = (long)N_ATOMS_C * NF;
  k_zero<<<(int)((ycount + 255) / 256), 256, 0, stream>>>(yacc, ycount);

  long rcount = (long)N_ATOMS_C * NB;
  k_cvt<<<(int)((rcount + 255) / 256), 256, 0, stream>>>(r, rbf, rcount);

  k_pack_w<<<(NB / 32) * 8, 32, 0, stream>>>(Wa, WaP);
  k_pack_w<<<(NG / 32) * 8, 32, 0, stream>>>(Wf2, Wf2P);
  k_pack_w<<<(NF / 32) * 8, 32, 0, stream>>>(W1, W1P);
  k_pack_w<<<(NB / 32) * 8, 32, 0, stream>>>(W2, W2P);

  int blocksAtoms = (N_ATOMS_C + 127) / 128;
  k_atomfilter<<<blocksAtoms, 256, 0, stream>>>(rbf, WaP, rf);

  k_edges<<<N_EDGES_C / 128, 256, 0, stream>>>(e, a2, Wf2P, bf2, rf, yacc);

  k_dense<<<blocksAtoms, 256, 0, stream>>>(yacc, W1P, b1, W2P, b2, out);
}
